// ContextEncoder_9259949490689
// MI455X (gfx1250) — matrix operand order repaired and checked by reading, not compiled, not run on hardware
//
#include <hip/hip_runtime.h>
#include <math.h>

// ---------------------------------------------------------------------------
// MI455X (gfx1250): fused GVP message-passing.
//  - all linear maps on v_wmma_f32_16x16x32_f16 (wave32, 16-row tiles)
//  - LDS pointers in addrspace(3), weights in addrspace(1)  (no flat ops,
//    no generic->LDS null-check cndmasks)
//  - edge gathers via GLOBAL_LOAD_ASYNC_TO_LDS_B128 (+ s_wait_asynccnt)
// ---------------------------------------------------------------------------

#define LDA_ 336   // f16 staging row stride (halves); >= 320, rows stay 16B aligned

typedef __attribute__((ext_vector_type(16))) _Float16 v16h;
typedef __attribute__((ext_vector_type(8)))  _Float16 v8h;
typedef __attribute__((ext_vector_type(8)))  float    v8f;
typedef _Float16 h16;

// address-space qualified types
typedef __attribute__((address_space(3))) _Float16 ls_h16;
typedef __attribute__((address_space(3))) float    ls_f32;
typedef __attribute__((address_space(3))) int      ls_i32;
typedef __attribute__((address_space(3))) v8h      ls_v8h;
typedef __attribute__((address_space(1))) _Float16 gb_h16;
typedef __attribute__((address_space(1))) float    gb_f32;
typedef __attribute__((address_space(1))) v8h      gb_v8h;

#define GH(p) ((const gb_h16*)(p))
#define GF(p) ((const gb_f32*)(p))

union H16u { v16h v; v8h h[2]; };

// D = A(16xK f16 LDS, stride LDA_) @ W^T ; W row-major f16 [ncols x kdim] in global.
// Output 16x16 tile at columns [ncol0, ncol0+16). A-fragment layout per CDNA5
// ISA 7.12.2 (16-bit A 16x32); B striping matches A: lanes 0-15 K=k0..k0+7,
// k0+16..k0+23; lanes 16-31 K=k0+8..k0+15, k0+24..k0+31.
__device__ __forceinline__ v8f wmma_k(const ls_h16* A, const gb_h16* W, int kdim,
                                      int ncol0, int lane, v8f acc) {
  const int hi = lane >> 4, lo = lane & 15;
  const ls_h16* ar = A + lo * LDA_;
  const gb_h16* wr = W + (size_t)(ncol0 + lo) * (size_t)kdim;
  for (int k0 = 0; k0 < kdim; k0 += 32) {
    H16u a, b;
    a.h[0] = *(const ls_v8h*)(ar + k0 + hi * 8);
    a.h[1] = *(const ls_v8h*)(ar + k0 + 16 + hi * 8);
    b.h[0] = *(const gb_v8h*)(wr + k0 + hi * 8);
    b.h[1] = *(const gb_v8h*)(wr + k0 + 16 + hi * 8);
    acc = __builtin_amdgcn_wmma_f32_16x16x32_f16(false, a.v, false, b.v,
                                                 (short)0, acc, false, false);
  }
  return acc;
}

// stage a 16x64 f32 LDS plane into staging cols [0,64)
__device__ __forceinline__ void stage64(ls_h16* A, const ls_f32* src, int lane) {
  for (int idx = lane; idx < 1024; idx += 32)
    A[(idx >> 6) * LDA_ + (idx & 63)] = (h16)src[idx];
}
// stage 16 x ncols f32 (row-contiguous, LDS) into staging at column col0
__device__ __forceinline__ void stageN(ls_h16* A, int col0, const ls_f32* src,
                                       int ncols, int lane) {
  for (int idx = lane; idx < 16 * ncols; idx += 32) {
    int m = idx / ncols, c = idx - m * ncols;
    A[m * LDA_ + col0 + c] = (h16)src[idx];
  }
}
// stage 16 x ncols f32 (row-contiguous, global) into staging at column col0
__device__ __forceinline__ void stageNg(ls_h16* A, int col0, const gb_f32* src,
                                        int ncols, int lane) {
  for (int idx = lane; idx < 16 * ncols; idx += 32) {
    int m = idx / ncols, c = idx - m * ncols;
    A[m * LDA_ + col0 + c] = (h16)src[idx];
  }
}
// 16x64 = A(16x64) @ W^T -> f32 LDS plane (no bias)
__device__ __forceinline__ void gemm64(const ls_h16* A, const gb_h16* W,
                                       ls_f32* dst, int lane) {
  const int hi = lane >> 4, lo = lane & 15;
  for (int nt = 0; nt < 4; nt++) {
    v8f acc = {};
    acc = wmma_k(A, W, 64, nt * 16, lane, acc);
#pragma unroll
    for (int r = 0; r < 8; r++)
      dst[(r + hi * 8) * 64 + nt * 16 + lo] = acc[r];
  }
}
// 16x64 = A(16x64) @ W^T + bias -> f32 LDS plane
__device__ __forceinline__ void gemm64b(const ls_h16* A, const gb_h16* W,
                                        ls_f32* dst, const gb_f32* bias, int lane) {
  const int hi = lane >> 4, lo = lane & 15;
  for (int nt = 0; nt < 4; nt++) {
    v8f acc = {};
    acc = wmma_k(A, W, 64, nt * 16, lane, acc);
    float bv = bias[nt * 16 + lo];
#pragma unroll
    for (int r = 0; r < 8; r++)
      dst[(r + hi * 8) * 64 + nt * 16 + lo] = acc[r] + bv;
  }
}

// GVLinear on a 16-row tile. Caller pre-stages scalar input (f16) at A cols
// [64, 64+sin). vin planes: 3 x (16x64) f32 LDS. Produces:
//   oS  = raw out_s (16 x sout, f32)  (also left as f16 at A[64..64+sout))
//   vwk = gated out_v planes
__device__ void gvlin16(ls_h16* A,
                        ls_f32* vi0, ls_f32* vi1, ls_f32* vi2,
                        ls_f32* vw0, ls_f32* vw1, ls_f32* vw2,
                        ls_f32* oS,
                        const gb_h16* Wv1, const gb_h16* Wv2, const gb_h16* Ws,
                        const gb_h16* Wg, const gb_f32* bg,
                        int sin, int sout, int lane) {
  ls_f32* vin[3] = { vi0, vi1, vi2 };
  ls_f32* vwk[3] = { vw0, vw1, vw2 };
  for (int d = 0; d < 3; d++) {            // vi = vec @ Wv1^T
    __syncthreads();
    stage64(A, vin[d], lane);
    __syncthreads();
    gemm64(A, Wv1, vwk[d], lane);
  }
  __syncthreads();
  for (int idx = lane; idx < 1024; idx += 32) {   // vnorm -> A[0,64)
    float a = vw0[idx], b = vw1[idx], c = vw2[idx];
    A[(idx >> 6) * LDA_ + (idx & 63)] = (h16)sqrtf(a * a + b * b + c * c);
  }
  __syncthreads();
  {                                         // out_s = [vnorm | sca] @ Ws^T
    const int kc = 64 + sin;
    const int hi = lane >> 4, lo = lane & 15;
    for (int nt = 0; nt < (sout >> 4); nt++) {
      v8f acc = {};
      acc = wmma_k(A, Ws, kc, nt * 16, lane, acc);
#pragma unroll
      for (int r = 0; r < 8; r++)
        oS[(r + hi * 8) * sout + nt * 16 + lo] = acc[r];
    }
  }
  __syncthreads();
  stageN(A, 64, oS, sout, lane);            // raw out_s f16 (gate input)
  for (int d = 0; d < 3; d++) {             // out_v = vi @ Wv2^T
    __syncthreads();
    stage64(A, vwk[d], lane);
    __syncthreads();
    gemm64(A, Wv2, vwk[d], lane);
  }
  __syncthreads();
  {                                         // gate = sigmoid(out_s @ Wg^T + bg)
    const int hi = lane >> 4, lo = lane & 15;
    for (int nt = 0; nt < 4; nt++) {
      v8f acc = {};
      acc = wmma_k(A + 64, Wg, sout, nt * 16, lane, acc);
#pragma unroll
      for (int r = 0; r < 8; r++) {
        int m = r + hi * 8, c = nt * 16 + lo;
        float g = 1.0f / (1.0f + __expf(-(acc[r] + bg[c])));
        vw0[m * 64 + c] *= g;
        vw1[m * 64 + c] *= g;
        vw2[m * 64 + c] *= g;
      }
    }
  }
  __syncthreads();
}

// async global->LDS 16B copy (CDNA5 TDM-adjacent path, ASYNCcnt-tracked)
__device__ __forceinline__ void async_b128(unsigned lds_addr, const void* gaddr) {
  asm volatile("global_load_async_to_lds_b128 %0, %1, off"
               :: "v"(lds_addr), "v"((unsigned long long)(uintptr_t)gaddr)
               : "memory");
}
__device__ __forceinline__ void wait_async0() {
  asm volatile("s_wait_asynccnt 0" ::: "memory");
}

// ---------------------------------------------------------------------------
// Kernels
// ---------------------------------------------------------------------------

__global__ void k_f32tof16(const float* s, h16* d, size_t n) {
  size_t i = (size_t)blockIdx.x * blockDim.x + threadIdx.x;
  if (i < n) ((gb_h16*)d)[i] = (h16)GF(s)[i];
}

__global__ void k_edge_pre(const float* pos, const float* ef, const int* ei, int E,
                           float* esf, float* unit, float* ann) {
  int e = blockIdx.x * blockDim.x + threadIdx.x;
  if (e >= E) return;
  int r = ei[e], c = ei[E + e];
  float dx = pos[r * 3 + 0] - pos[c * 3 + 0];
  float dy = pos[r * 3 + 1] - pos[c * 3 + 1];
  float dz = pos[r * 3 + 2] - pos[c * 3 + 2];
  float dist = sqrtf(dx * dx + dy * dy + dz * dz);
  float inv = 1.0f / (dist + 1e-7f);
  unit[(size_t)e * 3 + 0] = dx * inv;
  unit[(size_t)e * 3 + 1] = dy * inv;
  unit[(size_t)e * 3 + 2] = dz * inv;
  const float step = 10.0f / 59.0f;
  const float coeff = -0.5f / (step * step);
  float* o = esf + (size_t)e * 64;
  for (int g = 0; g < 60; g++) {
    float t = dist - step * (float)g;
    o[g] = __expf(coeff * t * t);
  }
  o[60] = ef[(size_t)e * 4 + 0]; o[61] = ef[(size_t)e * 4 + 1];
  o[62] = ef[(size_t)e * 4 + 2]; o[63] = ef[(size_t)e * 4 + 3];
  float a = 0.5f * (__cosf(dist * 0.31415926535f) + 1.0f);
  ann[e] = (dist <= 10.0f) ? a : 0.0f;
}

struct NodeArgs {
  const float* hs; const float* hv;
  float* outS; float* outV;
  const h16 *Wv1, *Wv2, *Ws, *Wg; const float* bg;
};

__global__ void __launch_bounds__(32)
k_node_gvl(NodeArgs na) {
  extern __shared__ char smem[];
  ls_h16* A   = (ls_h16*)smem;
  ls_f32* vin = (ls_f32*)(smem + 16 * LDA_ * 2);
  ls_f32* vwk = vin + 3072;
  ls_f32* oS  = vwk + 3072;
  const int lane = threadIdx.x;
  const int n0 = blockIdx.x * 16;
  for (int idx = lane; idx < 16 * 192; idx += 32) {
    int m = idx / 192, t = idx - m * 192, c = t / 3, d = t - c * 3;
    vin[d * 1024 + m * 64 + c] = GF(na.hv)[(size_t)n0 * 192 + idx];
  }
  stageNg(A, 64, GF(na.hs) + (size_t)n0 * 256, 256, lane);
  __syncthreads();
  gvlin16(A, vin, vin + 1024, vin + 2048, vwk, vwk + 1024, vwk + 2048, oS,
          GH(na.Wv1), GH(na.Wv2), GH(na.Ws), GH(na.Wg), GF(na.bg), 256, 256, lane);
  for (int idx = lane; idx < 16 * 256; idx += 32)
    ((gb_f32*)na.outS)[(size_t)n0 * 256 + idx] = oS[idx];
  for (int idx = lane; idx < 16 * 192; idx += 32) {
    int m = idx / 192, t = idx - m * 192, c = t / 3, d = t - c * 3;
    ((gb_f32*)na.outV)[(size_t)n0 * 192 + idx] = vwk[d * 1024 + m * 64 + c];
  }
}

struct EdgeArgs {
  const float* esf; const float* unitg; const float* anng;
  const int* rowg; const int* colg;
  const float* ns; const float* nv;
  float* aggrS; float* aggrV;
  const float* ve_w;
  const h16 *eWv1, *eWv2, *eWs, *eWg, *eDir, *n2eW, *e2nW, *scaW, *evnW,
            *oWv1, *oWv2, *oWs, *oWg;
  const float *e_bg, *n2e_b, *e2n_b, *sca_b, *evn_b, *o_bg;
};

__global__ void __launch_bounds__(32)
k_edge(EdgeArgs ea) {
  extern __shared__ char smem[];
  ls_h16* A    = (ls_h16*)smem;                    // 16 x 336 f16
  ls_f32* aS   = (ls_f32*)(smem + 16 * LDA_ * 2);  // 16x256 (ns_j -> y_s)
  ls_f32* nvj  = aS + 4096;                        // 16 x 192 (row-major, async dst)
  ls_f32* pX   = nvj + 3072;                       // 3 x 16x64
  ls_f32* pY   = pX + 3072;                        // 3 x 16x64
  ls_f32* sbA  = pY + 3072;                        // alpha
  ls_f32* sbB  = sbA + 1024;                       // beta
  ls_f32* oS   = sbB + 1024;                       // 16x256
  ls_f32* ann  = oS + 4096;                        // 16
  ls_f32* uu   = ann + 16;                         // 16x3
  ls_i32* rowi = (ls_i32*)(uu + 48);
  ls_i32* coli = rowi + 16;
  const int lane = threadIdx.x;
  const int e0 = blockIdx.x * 16;

  __builtin_prefetch(ea.oWs, 0, 0);
  __builtin_prefetch(ea.eWs, 0, 0);

  if (lane < 16) {
    int e = e0 + lane;
    rowi[lane] = GF(ea.rowg) ? 0 : 0;  // (placeholder removed below)
  }
  // meta loads
  if (lane < 16) {
    int e = e0 + lane;
    rowi[lane] = ((const __attribute__((address_space(1))) int*)ea.rowg)[e];
    coli[lane] = ((const __attribute__((address_space(1))) int*)ea.colg)[e];
    ann[lane]  = GF(ea.anng)[e];
    uu[lane * 3 + 0] = GF(ea.unitg)[(size_t)e * 3 + 0];
    uu[lane * 3 + 1] = GF(ea.unitg)[(size_t)e * 3 + 1];
    uu[lane * 3 + 2] = GF(ea.unitg)[(size_t)e * 3 + 2];
  }
  __syncthreads();
  // async gathers: ns_j (16 rows x 1KB) and nv_j (16 rows x 768B), 16B chunks
  {
    const unsigned aS_base  = (unsigned)(uintptr_t)aS;
    const unsigned nvj_base = (unsigned)(uintptr_t)nvj;
    for (int t = lane; t < 1024; t += 32) {       // ns_j
      int m = t >> 6, q = t & 63;
      async_b128(aS_base + (unsigned)(m * 1024 + q * 16),
                 ea.ns + (size_t)coli[m] * 256 + q * 4);
    }
    for (int t = lane; t < 768; t += 32) {        // nv_j
      int m = t / 48, q = t - m * 48;
      async_b128(nvj_base + (unsigned)(m * 768 + q * 16),
                 ea.nv + (size_t)coli[m] * 192 + q * 4);
    }
    wait_async0();
  }
  __syncthreads();
  // beta = ns_j @ n2e^T + b
  stageN(A, 0, aS, 256, lane);
  __syncthreads();
  {
    const int hi = lane >> 4, lo = lane & 15;
    for (int nt = 0; nt < 4; nt++) {
      v8f acc = {};
      acc = wmma_k(A, GH(ea.n2eW), 256, nt * 16, lane, acc);
      float bv = GF(ea.n2e_b)[nt * 16 + lo];
#pragma unroll
      for (int r = 0; r < 8; r++)
        sbB[(r + hi * 8) * 64 + nt * 16 + lo] = acc[r] + bv;
    }
  }
  __syncthreads();
  // edge_vec_feat planes; esf staged at A[64..128)
  for (int idx = lane; idx < 1024; idx += 32) {
    int m = idx >> 6, c = idx & 63;
    float vw = GF(ea.ve_w)[c];
    pX[idx]        = vw * uu[m * 3 + 0];
    pX[1024 + idx] = vw * uu[m * 3 + 1];
    pX[2048 + idx] = vw * uu[m * 3 + 2];
    A[m * LDA_ + 64 + c] = (h16)GF(ea.esf)[(size_t)(e0 + m) * 64 + c];
  }
  __syncthreads();
  // egvp GVLinear
  gvlin16(A, pX, pX + 1024, pX + 2048, pY, pY + 1024, pY + 2048, oS,
          GH(ea.eWv1), GH(ea.eWv2), GH(ea.eWs), GH(ea.eWg), GF(ea.e_bg),
          64, 64, lane);
  // pY = ev (gated), oS[0..1024) = raw out_s0
  for (int d = 0; d < 3; d++) {                   // vn_leaky dir GEMMs
    __syncthreads();
    stage64(A, pY + d * 1024, lane);
    __syncthreads();
    gemm64(A, GH(ea.eDir), pX + d * 1024, lane);
  }
  __syncthreads();
  for (int idx = lane; idx < 1024; idx += 32) {
    float x0 = pY[idx], x1 = pY[1024 + idx], x2 = pY[2048 + idx];
    float d0 = pX[idx], d1 = pX[1024 + idx], d2 = pX[2048 + idx];
    float dot = x0 * d0 + x1 * d1 + x2 * d2;
    float dns = d0 * d0 + d1 * d1 + d2 * d2;
    float t = dot / (dns + 1e-6f);
    float msk = (dot >= 0.0f) ? 1.0f : 0.0f;
    pY[idx]        = 0.2f * x0 + 0.8f * (msk * x0 + (1.0f - msk) * (x0 - t * d0));
    pY[1024 + idx] = 0.2f * x1 + 0.8f * (msk * x1 + (1.0f - msk) * (x1 - t * d1));
    pY[2048 + idx] = 0.2f * x2 + 0.8f * (msk * x2 + (1.0f - msk) * (x2 - t * d2));
  }
  __syncthreads();
  for (int idx = lane; idx < 1024; idx += 32) {   // es = leaky(out_s0, .01)
    float v = oS[idx];
    v = (v >= 0.0f) ? v : 0.01f * v;
    A[(idx >> 6) * LDA_ + (idx & 63)] = (h16)v;
  }
  __syncthreads();
  gemm64b(A, GH(ea.e2nW), sbA, GF(ea.e2n_b), lane);   // alpha
  {                                               // y_s = ns_j*(es@scaW^T+b)
    const int hi = lane >> 4, lo = lane & 15;
    for (int nt = 0; nt < 16; nt++) {
      v8f acc = {};
      acc = wmma_k(A, GH(ea.scaW), 64, nt * 16, lane, acc);
      float bv = GF(ea.sca_b)[nt * 16 + lo];
#pragma unroll
      for (int r = 0; r < 8; r++) {
        int m = r + hi * 8, n = nt * 16 + lo;
        float v = (acc[r] + bv) * aS[m * 256 + n];
        aS[m * 256 + n] = v;
        A[m * LDA_ + 64 + n] = (h16)v;
      }
    }
  }
  for (int d = 0; d < 3; d++) {                   // evn = evnW@ev' + b
    __syncthreads();
    stage64(A, pY + d * 1024, lane);
    __syncthreads();
    gemm64b(A, GH(ea.evnW), pX + d * 1024, GF(ea.evn_b), lane);
  }
  __syncthreads();
  for (int idx = lane; idx < 1024; idx += 32) {   // y_v = alpha*nv_j + beta*evn
    int m = idx >> 6, c = idx & 63;
    float a = sbA[idx], b = sbB[idx];
    pY[idx]        = a * nvj[m * 192 + c * 3 + 0] + b * pX[idx];
    pY[1024 + idx] = a * nvj[m * 192 + c * 3 + 1] + b * pX[1024 + idx];
    pY[2048 + idx] = a * nvj[m * 192 + c * 3 + 2] + b * pX[2048 + idx];
  }
  __syncthreads();
  // mm_out GVLinear (scalar input y_s already staged at A[64..320))
  gvlin16(A, pY, pY + 1024, pY + 2048, pX, pX + 1024, pX + 2048, oS,
          GH(ea.oWv1), GH(ea.oWv2), GH(ea.oWs), GH(ea.oWg), GF(ea.o_bg),
          256, 256, lane);
  // scatter with anneal
  for (int idx = lane; idx < 16 * 256; idx += 32) {
    int m = idx >> 8;
    atomicAdd(ea.aggrS + (size_t)rowi[m] * 256 + (idx & 255), oS[idx] * ann[m]);
  }
  for (int idx = lane; idx < 1024; idx += 32) {
    int m = idx >> 6, c = idx & 63;
    float an = ann[m];
    size_t base = (size_t)rowi[m] * 192 + c * 3;
    atomicAdd(ea.aggrV + base + 0, pX[idx] * an);
    atomicAdd(ea.aggrV + base + 1, pX[1024 + idx] * an);
    atomicAdd(ea.aggrV + base + 2, pX[2048 + idx] * an);
  }
}

struct PostArgs {
  const float *cs, *cv, *agS, *agV;
  const float *lnsw, *lnsb, *lnvw, *lnvb;
  const h16 *dirW, *Wv1, *Wv2, *Ws, *Wg; const float* bg;
  float *hs, *hv;
};

__global__ void __launch_bounds__(32)
k_node_post(PostArgs pa) {
  extern __shared__ char smem[];
  ls_h16* A   = (ls_h16*)smem;
  ls_f32* vin = (ls_f32*)(smem + 16 * LDA_ * 2);
  ls_f32* vwk = vin + 3072;
  ls_f32* oS  = vwk + 3072;
  ls_f32* mu  = oS + 4096;
  ls_f32* rs  = mu + 16;
  const int lane = threadIdx.x;
  const int n0 = blockIdx.x * 16;
  for (int idx = lane; idx < 16 * 256; idx += 32)
    oS[idx] = GF(pa.cs)[(size_t)n0 * 256 + idx] + GF(pa.agS)[(size_t)n0 * 256 + idx];
  __syncthreads();
  if (lane < 16) {
    float s = 0.0f;
    for (int k = 0; k < 256; k++) s += oS[lane * 256 + k];
    float m_ = s * (1.0f / 256.0f);
    float v = 0.0f;
    for (int k = 0; k < 256; k++) { float d = oS[lane * 256 + k] - m_; v += d * d; }
    mu[lane] = m_;
    rs[lane] = rsqrtf(v * (1.0f / 256.0f) + 1e-5f);
  }
  __syncthreads();
  for (int idx = lane; idx < 16 * 256; idx += 32) {
    int m = idx >> 8, k = idx & 255;
    float v = (oS[idx] - mu[m]) * rs[m] * GF(pa.lnsw)[k] + GF(pa.lnsb)[k];
    v = (v >= 0.0f) ? v : 0.01f * v;
    A[m * LDA_ + 64 + k] = (h16)v;
  }
  for (int idx = lane; idx < 16 * 192; idx += 32) {
    int m = idx / 192, t = idx - m * 192, c = t / 3, d = t - c * 3;
    vin[d * 1024 + m * 64 + c] =
        GF(pa.cv)[(size_t)n0 * 192 + idx] + GF(pa.agV)[(size_t)n0 * 192 + idx];
  }
  __syncthreads();
  if (lane < 16) {
    float s = 0.0f;
    for (int c = 0; c < 64; c++)
      s += vin[lane * 64 + c] + vin[1024 + lane * 64 + c] + vin[2048 + lane * 64 + c];
    float m_ = s * (1.0f / 192.0f);
    float v = 0.0f;
    for (int c = 0; c < 64; c++) {
      float a = vin[lane * 64 + c] - m_, b = vin[1024 + lane * 64 + c] - m_,
            d = vin[2048 + lane * 64 + c] - m_;
      v += a * a + b * b + d * d;
    }
    mu[lane] = m_;
    rs[lane] = rsqrtf(v * (1.0f / 192.0f) + 1e-5f);
  }
  __syncthreads();
  for (int idx = lane; idx < 1024; idx += 32) {
    int m = idx >> 6, c = idx & 63;
#pragma unroll
    for (int d = 0; d < 3; d++)
      vin[d * 1024 + idx] = (vin[d * 1024 + idx] - mu[m]) * rs[m] *
                            GF(pa.lnvw)[c * 3 + d] + GF(pa.lnvb)[c * 3 + d];
  }
  __syncthreads();
  for (int d = 0; d < 3; d++) {                   // vn_leaky(blk_dir)
    __syncthreads();
    stage64(A, vin + d * 1024, lane);
    __syncthreads();
    gemm64(A, GH(pa.dirW), vwk + d * 1024, lane);
  }
  __syncthreads();
  for (int idx = lane; idx < 1024; idx += 32) {
    float x0 = vin[idx], x1 = vin[1024 + idx], x2 = vin[2048 + idx];
    float d0 = vwk[idx], d1 = vwk[1024 + idx], d2 = vwk[2048 + idx];
    float dot = x0 * d0 + x1 * d1 + x2 * d2;
    float dns = d0 * d0 + d1 * d1 + d2 * d2;
    float t = dot / (dns + 1e-6f);
    float msk = (dot >= 0.0f) ? 1.0f : 0.0f;
    vin[idx]        = 0.2f * x0 + 0.8f * (msk * x0 + (1.0f - msk) * (x0 - t * d0));
    vin[1024 + idx] = 0.2f * x1 + 0.8f * (msk * x1 + (1.0f - msk) * (x1 - t * d1));
    vin[2048 + idx] = 0.2f * x2 + 0.8f * (msk * x2 + (1.0f - msk) * (x2 - t * d2));
  }
  __syncthreads();
  gvlin16(A, vin, vin + 1024, vin + 2048, vwk, vwk + 1024, vwk + 2048, oS,
          GH(pa.Wv1), GH(pa.Wv2), GH(pa.Ws), GH(pa.Wg), GF(pa.bg), 256, 256, lane);
  for (int idx = lane; idx < 16 * 256; idx += 32)
    ((gb_f32*)pa.hs)[(size_t)n0 * 256 + idx] += oS[idx];
  for (int idx = lane; idx < 16 * 192; idx += 32) {
    int m = idx / 192, t = idx - m * 192, c = t / 3, d = t - c * 3;
    ((gb_f32*)pa.hv)[(size_t)n0 * 192 + idx] += vwk[d * 1024 + m * 64 + c];
  }
}

// ---------------------------------------------------------------------------
// Host launch
// ---------------------------------------------------------------------------

extern "C" void kernel_launch(void* const* d_in, const int* in_sizes, int n_in,
                              void* d_out, int out_size, void* d_ws, size_t ws_size,
                              hipStream_t stream) {
  (void)out_size; (void)ws_size;
  const int N_ = 4096, E_ = 131072, L_ = 6;
  const float* node_sca = (const float*)d_in[0];
  const float* node_vec = (const float*)d_in[1];
  const float* pos      = (const float*)d_in[2];
  const float* edge_feature = (const float*)d_in[3];
  const int*   edge_index   = (const int*)d_in[n_in - 1];

  char* ws = (char*)d_ws;
  size_t off = 0;
  auto alloc = [&](size_t bytes) -> void* {
    off = (off + 255) & ~(size_t)255;
    void* p = ws + off; off += bytes; return p;
  };

  static const int cvt[] = {4, 5, 6, 7, 8, 14, 16, 17, 18, 19, 21, 22, 24,
                            26, 27, 28, 29, 31, 32, 33, 34, 36, 38, 39, 40, 41};
  h16* w16[64] = {};
  for (int t = 0; t < (int)(sizeof(cvt) / sizeof(int)); t++) {
    int i = cvt[t];
    size_t n = (size_t)in_sizes[i];
    w16[i] = (h16*)alloc(n * sizeof(h16));
    k_f32tof16<<<(int)((n + 255) / 256), 256, 0, stream>>>((const float*)d_in[i], w16[i], n);
  }

  float* esf   = (float*)alloc((size_t)E_ * 64 * 4);
  float* unitb = (float*)alloc((size_t)E_ * 3 * 4);
  float* annb  = (float*)alloc((size_t)E_ * 4);
  float* ns    = (float*)alloc((size_t)N_ * 256 * 4);
  float* nv    = (float*)alloc((size_t)N_ * 192 * 4);
  float* cs    = (float*)alloc((size_t)N_ * 256 * 4);
  float* cv    = (float*)alloc((size_t)N_ * 192 * 4);
  float* agS   = (float*)alloc((size_t)N_ * 256 * 4);
  float* agV   = (float*)alloc((size_t)N_ * 192 * 4);

  float* hs = (float*)d_out;
  float* hv = hs + (size_t)N_ * 256;
  hipMemcpyAsync(hs, node_sca, (size_t)N_ * 256 * 4, hipMemcpyDeviceToDevice, stream);
  hipMemcpyAsync(hv, node_vec, (size_t)N_ * 192 * 4, hipMemcpyDeviceToDevice, stream);

  k_edge_pre<<<E_ / 256, 256, 0, stream>>>(pos, edge_feature, edge_index, E_,
                                           esf, unitb, annb);

  const size_t LDS_N = (size_t)16 * LDA_ * 2 + (size_t)(3072 + 3072 + 4096) * 4;
  const size_t LDS_P = LDS_N + 32 * 4;
  const size_t LDS_E = (size_t)16 * LDA_ * 2 +
      (size_t)(4096 + 3 * 3072 + 2 * 1024 + 4096 + 16 + 48) * 4 + 32 * 4;

  for (int l = 0; l < L_; l++) {
    auto Wh = [&](int i, size_t per) -> const h16*   { return w16[i] + (size_t)l * per; };
    auto Bf = [&](int i, size_t per) -> const float* { return (const float*)d_in[i] + (size_t)l * per; };

    hipMemsetAsync(agS, 0, (size_t)N_ * 256 * 4, stream);
    hipMemsetAsync(agV, 0, (size_t)N_ * 192 * 4, stream);

    NodeArgs a1{hs, hv, ns, nv, Wh(28, 4096), Wh(29, 4096), Wh(27, 81920),
                Wh(26, 16384), Bf(30, 64)};                         // mm_node
    k_node_gvl<<<N_ / 16, 32, LDS_N, stream>>>(a1);
    NodeArgs a2{hs, hv, cs, cv, Wh(7, 4096), Wh(8, 4096), Wh(6, 81920),
                Wh(5, 16384), Bf(9, 64)};                           // cent
    k_node_gvl<<<N_ / 16, 32, LDS_N, stream>>>(a2);

    EdgeArgs ea{esf, unitb, annb, edge_index, edge_index + E_,
                ns, nv, agS, agV, Bf(43, 64),
                Wh(18, 4096), Wh(19, 4096), Wh(17, 8192), Wh(16, 4096), Wh(21, 4096),
                Wh(24, 16384), Wh(14, 4096), Wh(36, 16384), Wh(22, 4096),
                Wh(33, 4096), Wh(34, 4096), Wh(32, 81920), Wh(31, 16384),
                Bf(20, 64), Bf(25, 64), Bf(15, 64), Bf(37, 256), Bf(23, 64), Bf(35, 64)};
    k_edge<<<E_ / 16, 32, LDS_E, stream>>>(ea);

    PostArgs pa{cs, cv, agS, agV,
                Bf(11, 256), Bf(10, 256), Bf(13, 192), Bf(12, 192),
                Wh(4, 4096), Wh(40, 4096), Wh(41, 4096), Wh(39, 81920),
                Wh(38, 16384), Bf(42, 64),
                hs, hv};
    k_node_post<<<N_ / 16, 32, LDS_P, stream>>>(pa);
  }
}
